// GET_Model_223338299542
// MI455X (gfx1250) — compile-verified, matrix operand order checked by reading, not run on hardware
//
#include <hip/hip_runtime.h>
#include <hip/hip_bf16.h>

// ---------------------------------------------------------------------------
// Types for CDNA5 WMMA (wave32): bf16 16x16x32, f32 accumulate
// ---------------------------------------------------------------------------
typedef __attribute__((ext_vector_type(16))) __bf16   v16bf;
typedef __attribute__((ext_vector_type(8)))  float    v8f;
typedef __attribute__((ext_vector_type(4)))  unsigned v4u;

union V16 { v16bf bf; v4u q[2]; };

__device__ __forceinline__ unsigned short f2bf_u(float f) {
    union { float f; unsigned u; } c; c.f = f;
    unsigned r = c.u + 0x7FFFu + ((c.u >> 16) & 1u);   // round-to-nearest-even
    return (unsigned short)(r >> 16);
}

__device__ __forceinline__ float sigmoidf_(float x) { return 1.f / (1.f + expf(-x)); }

// ---------------------------------------------------------------------------
// Operand packing: fp32 -> bf16, K padded to Kp (multiple of 32), zero-filled.
// A: row-major (M x K, lda)      -> Ap[bat*M + m][Kp]
// B: row-major (K x N, ldb)      -> Bp[bat*N + n][Kp]   (i.e. column-major bf16)
// Each thread emits one packed uint (2 bf16).
// ---------------------------------------------------------------------------
__global__ void k_pack_a(const float* __restrict__ A, int lda, long long sA,
                         unsigned* __restrict__ Ap, int M, int K, int Kp,
                         long long total /* batch*M*Kp/2 */)
{
    long long i = (long long)blockIdx.x * blockDim.x + threadIdx.x;
    if (i >= total) return;
    const int Kp2 = Kp >> 1;
    int kk  = (int)(i % Kp2) * 2;
    long long t = i / Kp2;
    int m   = (int)(t % M);
    int bat = (int)(t / M);
    const float* Ab = A + (long long)bat * sA + (long long)m * lda;
    unsigned lo = (kk     < K) ? f2bf_u(Ab[kk])     : 0u;
    unsigned hi = (kk + 1 < K) ? f2bf_u(Ab[kk + 1]) : 0u;
    Ap[i] = lo | (hi << 16);
}

__global__ void k_pack_b(const float* __restrict__ B, int ldb, long long sB,
                         unsigned* __restrict__ Bp, int N, int K, int Kp,
                         long long total /* batch*N*Kp/2 */)
{
    long long i = (long long)blockIdx.x * blockDim.x + threadIdx.x;
    if (i >= total) return;
    const int Kp2 = Kp >> 1;
    int kk  = (int)(i % Kp2) * 2;
    long long t = i / Kp2;
    int n   = (int)(t % N);
    int bat = (int)(t / N);
    const float* Bb = B + (long long)bat * sB + n;
    unsigned lo = (kk     < K) ? f2bf_u(Bb[(long long)kk * ldb])       : 0u;
    unsigned hi = (kk + 1 < K) ? f2bf_u(Bb[(long long)(kk + 1) * ldb]) : 0u;
    Bp[i] = lo | (hi << 16);
}

// ---------------------------------------------------------------------------
// WMMA GEMM on packed bf16 operands.
// Register-blocked: each wave computes a 32x32 C macro-tile (2x2 WMMA tiles),
// 4 waves/block stacked along M (block covers 128 rows x 32 cols).
// Per 16x16x32 k-step: 2 A-fragments + 2 B-fragments (4x b128 each pair)
// feed 4 WMMAs -> 2x operand reuse vs one-tile-per-wave.
// Branch-free K loop (K zero-padded by packing); M/N edges: clamp load
// indices (in-bounds garbage only pollutes store-masked outputs).
// act: 0 = none, 1 = relu
// ---------------------------------------------------------------------------
__global__ void wmma_gemm_kernel(const unsigned short* __restrict__ Ap,
                                 const unsigned short* __restrict__ Bp,
                                 float* __restrict__ C, int ldc, long long sC,
                                 const float* __restrict__ bias,
                                 int M, int N, int Kp, int accum, int act)
{
    const int bat = blockIdx.z;
    Ap += (long long)bat * M * Kp;
    Bp += (long long)bat * N * Kp;
    C  += (long long)bat * sC;

    const int lane = threadIdx.x & 31;
    const int wave = threadIdx.x >> 5;
    const int n0   = blockIdx.x * 32;
    const int m0   = (blockIdx.y * 4 + wave) * 32;
    if (m0 >= M) return;                           // wave-uniform exit

    const int khalf = lane >> 4;
    const int l15   = lane & 15;
    const int mr0   = min(m0 + l15,      M - 1);   // clamped load rows
    const int mr1   = min(m0 + 16 + l15, M - 1);
    const int nc0s  = n0 + l15;                    // store columns
    const int nc1s  = n0 + 16 + l15;
    const int nc0   = min(nc0s, N - 1);            // clamped load columns
    const int nc1   = min(nc1s, N - 1);

    const unsigned short* arow0 = Ap + (long long)mr0 * Kp + khalf * 8;
    const unsigned short* arow1 = Ap + (long long)mr1 * Kp + khalf * 8;
    const unsigned short* bcol0 = Bp + (long long)nc0 * Kp + khalf * 8;
    const unsigned short* bcol1 = Bp + (long long)nc1 * Kp + khalf * 8;

    v8f acc00 = {}, acc01 = {}, acc10 = {}, acc11 = {};
#pragma unroll 2
    for (int k0 = 0; k0 < Kp; k0 += 32) {
        V16 a0, a1, b0, b1;
        a0.q[0] = *(const v4u*)(arow0 + k0);
        a0.q[1] = *(const v4u*)(arow0 + k0 + 16);
        a1.q[0] = *(const v4u*)(arow1 + k0);
        a1.q[1] = *(const v4u*)(arow1 + k0 + 16);
        b0.q[0] = *(const v4u*)(bcol0 + k0);
        b0.q[1] = *(const v4u*)(bcol0 + k0 + 16);
        b1.q[0] = *(const v4u*)(bcol1 + k0);
        b1.q[1] = *(const v4u*)(bcol1 + k0 + 16);
        acc00 = __builtin_amdgcn_wmma_f32_16x16x32_bf16(false, a0.bf, false, b0.bf,
                                                        (short)0, acc00, false, false);
        acc01 = __builtin_amdgcn_wmma_f32_16x16x32_bf16(false, a0.bf, false, b1.bf,
                                                        (short)0, acc01, false, false);
        acc10 = __builtin_amdgcn_wmma_f32_16x16x32_bf16(false, a1.bf, false, b0.bf,
                                                        (short)0, acc10, false, false);
        acc11 = __builtin_amdgcn_wmma_f32_16x16x32_bf16(false, a1.bf, false, b1.bf,
                                                        (short)0, acc11, false, false);
    }

    const float bv0 = bias ? bias[nc0] : 0.f;
    const float bv1 = bias ? bias[nc1] : 0.f;
    const bool  n0ok = (nc0s < N);
    const bool  n1ok = (nc1s < N);
#pragma unroll
    for (int v = 0; v < 8; ++v) {
        const int r0 = m0 + v + 8 * khalf;        // C layout: VGPR v -> row v / v+8
        const int r1 = r0 + 16;
        if (r0 < M && n0ok) {
            long long ci = (long long)r0 * ldc + nc0s;
            float o = acc00[v] + bv0 + (accum ? C[ci] : 0.f);
            if (act == 1) o = fmaxf(o, 0.f);
            C[ci] = o;
        }
        if (r0 < M && n1ok) {
            long long ci = (long long)r0 * ldc + nc1s;
            float o = acc01[v] + bv1 + (accum ? C[ci] : 0.f);
            if (act == 1) o = fmaxf(o, 0.f);
            C[ci] = o;
        }
        if (r1 < M && n0ok) {
            long long ci = (long long)r1 * ldc + nc0s;
            float o = acc10[v] + bv0 + (accum ? C[ci] : 0.f);
            if (act == 1) o = fmaxf(o, 0.f);
            C[ci] = o;
        }
        if (r1 < M && n1ok) {
            long long ci = (long long)r1 * ldc + nc1s;
            float o = acc11[v] + bv1 + (accum ? C[ci] : 0.f);
            if (act == 1) o = fmaxf(o, 0.f);
            C[ci] = o;
        }
    }
}

// ---------------------------------------------------------------------------
// Elementwise / small kernels
// ---------------------------------------------------------------------------
__global__ void k_gather(const int* __restrict__ idx, const float* __restrict__ emb,
                         float* __restrict__ out, long long tokens, int D)
{
    long long i = (long long)blockIdx.x * blockDim.x + threadIdx.x;
    if (i >= tokens * D) return;
    long long t = i / D; int d = (int)(i % D);
    out[i] = emb[(long long)idx[t] * D + d];
}

__global__ void k_sigmoid(float* io, long long n) {
    long long i = (long long)blockIdx.x * blockDim.x + threadIdx.x;
    if (i < n) io[i] = sigmoidf_(io[i]);
}

__global__ void k_mul_inplace(float* io, const float* __restrict__ x, long long n) {
    long long i = (long long)blockIdx.x * blockDim.x + threadIdx.x;
    if (i < n) io[i] *= x[i];
}

// out = tanh(h)*z + x*(1-z)
__global__ void k_ggnn_out(const float* __restrict__ h, const float* __restrict__ z,
                           const float* __restrict__ x, float* __restrict__ out, long long n)
{
    long long i = (long long)blockIdx.x * blockDim.x + threadIdx.x;
    if (i >= n) return;
    float zz = z[i];
    out[i] = tanhf(h[i]) * zz + x[i] * (1.f - zz);
}

// io[i] = tanh(io[i] + left[(row/rowsPerLeft)*D + col])
__global__ void k_add_tanh_bcast(float* io, const float* __restrict__ left,
                                 int D, int rowsPerLeft, long long n)
{
    long long i = (long long)blockIdx.x * blockDim.x + threadIdx.x;
    if (i >= n) return;
    long long r = i / D; int d = (int)(i % D);
    io[i] = tanhf(io[i] + left[(r / rowsPerLeft) * (long long)D + d]);
}

// per-(b,i): out[b*L+i] = sum_j adj[b,i,j] * x[b*L+j]
__global__ void k_adj_matvec(const float* __restrict__ adj, const float* __restrict__ x,
                             float* __restrict__ out, int batch, int L)
{
    int i = blockIdx.x * blockDim.x + threadIdx.x;
    if (i >= batch * L) return;
    int b = i / L, r = i % L;
    const float* arow = adj + ((long long)b * L + r) * L;
    const float* xb   = x + (long long)b * L;
    float s = 0.f;
    for (int j = 0; j < L; ++j) s += arow[j] * xb[j];
    out[i] = s;
}

// GGNN with scalar hidden size (dout == 1): all gate matmuls are scalars
__global__ void k_scorer_gates(const float* __restrict__ a, const float* __restrict__ x,
    const float* wz0, const float* bz0, const float* wz1, const float* bz1,
    const float* wr0, const float* br0, const float* wr1, const float* br1,
    const float* wh0, const float* bh0, const float* wh1, const float* bh1,
    float* __restrict__ out, int n)
{
    int i = blockIdx.x * blockDim.x + threadIdx.x;
    if (i >= n) return;
    float av = a[i], xv = x[i];
    float z = sigmoidf_(av * wz0[0] + bz0[0] + xv * wz1[0] + bz1[0]);
    float r = sigmoidf_(av * wr0[0] + br0[0] + xv * wr1[0] + br1[0]);
    float h = tanhf(av * wh0[0] + bh0[0] + r * xv * wh1[0] + bh1[0]);
    out[i] = h * z + xv * (1.f - z);
}

// GSL: per doc, keep top-`keep` scores (stable rank, first-index ties like top_k),
// mask = OR of row/col keep flags, adj_ref = adj * mask
__global__ void k_gsl(const float* __restrict__ adj, const float* __restrict__ score,
                      float* __restrict__ out, int L, int keep)
{
    int b = blockIdx.x;
    __shared__ float s[128];
    __shared__ float keepf[128];
    const float* sc = score + (long long)b * L;
    for (int i = threadIdx.x; i < L; i += blockDim.x) s[i] = sc[i];
    __syncthreads();
    for (int i = threadIdx.x; i < L; i += blockDim.x) {
        float si = s[i];
        int rank = 0;
        for (int j = 0; j < L; ++j) {
            float sj = s[j];
            rank += (sj > si) || (sj == si && j < i);
        }
        keepf[i] = (rank < keep) ? 1.f : 0.f;
    }
    __syncthreads();
    const float* ab = adj + (long long)b * L * L;
    float*       ob = out + (long long)b * L * L;
    for (int e = threadIdx.x; e < L * L; e += blockDim.x) {
        int r = e / L, c = e % L;
        ob[e] = ab[e] * fmaxf(keepf[r], keepf[c]);
    }
}

// qr[b,d] = sum_l x[b,l,d]*mask[b,l] / sum_l mask[b,l]
__global__ void k_masked_mean(const float* __restrict__ x, const float* __restrict__ mask,
                              float* __restrict__ out, int B, int L, int D)
{
    int i = blockIdx.x * blockDim.x + threadIdx.x;
    if (i >= B * D) return;
    int b = i / D, d = i % D;
    float s = 0.f, ms = 0.f;
    for (int l = 0; l < L; ++l) {
        float m = mask[b * L + l];
        s  += x[((long long)b * L + l) * D + d] * m;
        ms += m;
    }
    out[i] = s / ms;
}

// softmax over axis L of logits (nb, L, H); mask (nb, L) optional (0 -> -inf)
__global__ void k_softmax_axis(const float* __restrict__ logits, const float* __restrict__ mask,
                               float* __restrict__ out, int nb, int L, int H)
{
    int i = blockIdx.x * blockDim.x + threadIdx.x;
    if (i >= nb * H) return;
    int b = i / H, h = i % H;
    const float* lg = logits + (long long)b * L * H + h;
    const float* mk = mask ? (mask + (long long)b * L) : nullptr;
    float mx = -1e30f;
    for (int l = 0; l < L; ++l) {
        bool on = (!mk) || (mk[l] != 0.f);
        if (on) mx = fmaxf(mx, lg[(long long)l * H]);
    }
    float sum = 0.f;
    for (int l = 0; l < L; ++l) {
        bool on = (!mk) || (mk[l] != 0.f);
        sum += on ? expf(lg[(long long)l * H] - mx) : 0.f;
    }
    float inv = 1.f / sum;
    for (int l = 0; l < L; ++l) {
        bool on = (!mk) || (mk[l] != 0.f);
        out[(long long)b * L * H + (long long)l * H + h] =
            on ? expf(lg[(long long)l * H] - mx) * inv : 0.f;
    }
}

// out[b, d*H + h] = sum_l right[b,l,d] * w[b,l,h]   (einsum bld,blh->bdh, flattened)
__global__ void k_attend(const float* __restrict__ right, const float* __restrict__ w,
                         float* __restrict__ out, int nb, int L, int D, int H)
{
    long long i = (long long)blockIdx.x * blockDim.x + threadIdx.x;
    long long total = (long long)nb * D * H;
    if (i >= total) return;
    int h = (int)(i % H);
    long long t = i / H;
    int d = (int)(t % D);
    int b = (int)(t / D);
    float s = 0.f;
    for (int l = 0; l < L; ++l)
        s += right[((long long)b * L + l) * D + d] * w[((long long)b * L + l) * H + h];
    out[(long long)b * D * H + (long long)d * H + h] = s;
}

__global__ void k_concat(const float* __restrict__ a, int da,
                         const float* __restrict__ b, int db,
                         float* __restrict__ out, int rows)
{
    long long i = (long long)blockIdx.x * blockDim.x + threadIdx.x;
    int dt = da + db;
    if (i >= (long long)rows * dt) return;
    int r = (int)(i / dt), c = (int)(i % dt);
    out[i] = (c < da) ? a[(long long)r * da + c] : b[(long long)r * db + (c - da)];
}

__global__ void k_matmul_small(const float* __restrict__ A, const float* __restrict__ W,
                               const float* __restrict__ bias, float* __restrict__ out,
                               int M, int K, int N)
{
    int i = blockIdx.x * blockDim.x + threadIdx.x;
    if (i >= M * N) return;
    int m = i / N, n = i % N;
    float s = bias ? bias[n] : 0.f;
    for (int k = 0; k < K; ++k) s += A[(long long)m * K + k] * W[(long long)k * N + n];
    out[i] = s;
}

// ---------------------------------------------------------------------------
// Host-side helpers
// ---------------------------------------------------------------------------
struct GG {
    const float *proj, *wz0, *bz0, *wz1, *bz1, *wr0, *br0, *wr1, *br1, *wh0, *bh0, *wh1, *bh1;
};

static inline dim3 ewg(long long n) { return dim3((unsigned)((n + 255) / 256)); }

struct GemmCtx { unsigned short *Ap, *Bp; };

static void gemm(hipStream_t st, const GemmCtx& cx,
                 const float* A, int lda, long long sA,
                 const float* B, int ldb, long long sB,
                 float* C, int ldc, long long sC,
                 const float* bias, int M, int N, int K,
                 int batch, int accum, int act)
{
    const int Kp = (K + 31) & ~31;
    const long long na2 = (long long)batch * M * Kp / 2;
    const long long nb2 = (long long)batch * N * Kp / 2;
    k_pack_a<<<ewg(na2), 256, 0, st>>>(A, lda, sA, (unsigned*)cx.Ap, M, K, Kp, na2);
    k_pack_b<<<ewg(nb2), 256, 0, st>>>(B, ldb, sB, (unsigned*)cx.Bp, N, K, Kp, nb2);
    dim3 grid((N + 31) / 32, (M + 127) / 128, batch);
    wmma_gemm_kernel<<<grid, 128, 0, st>>>(cx.Ap, cx.Bp, C, ldc, sC,
                                           bias, M, N, Kp, accum, act);
}

static void ggnn_big(hipStream_t st, const GemmCtx& cx, const GG& g,
                     const float* adj, int batch, int L,
                     const float* x, int din,
                     float* X, float* Ag, float* T1, float* T2, float* T4, float* out)
{
    const int M = batch * L;
    const long long n = (long long)M * 300;
    // x projected
    gemm(st, cx, x, din, 0, g.proj, 300, 0, X, 300, 0, nullptr, M, 300, din, 1, 0, 0);
    // a = adj @ X  (batched)
    gemm(st, cx, adj, L, (long long)L * L, X, 300, (long long)L * 300,
         Ag, 300, (long long)L * 300, nullptr, L, 300, L, batch, 0, 0);
    // z
    gemm(st, cx, Ag, 300, 0, g.wz0, 300, 0, T1, 300, 0, g.bz0, M, 300, 300, 1, 0, 0);
    gemm(st, cx, X,  300, 0, g.wz1, 300, 0, T1, 300, 0, g.bz1, M, 300, 300, 1, 1, 0);
    k_sigmoid<<<ewg(n), 256, 0, st>>>(T1, n);
    // r
    gemm(st, cx, Ag, 300, 0, g.wr0, 300, 0, T2, 300, 0, g.br0, M, 300, 300, 1, 0, 0);
    gemm(st, cx, X,  300, 0, g.wr1, 300, 0, T2, 300, 0, g.br1, M, 300, 300, 1, 1, 0);
    k_sigmoid<<<ewg(n), 256, 0, st>>>(T2, n);
    k_mul_inplace<<<ewg(n), 256, 0, st>>>(T2, X, n);   // T2 = r * x
    // h
    gemm(st, cx, Ag, 300, 0, g.wh0, 300, 0, T4, 300, 0, g.bh0, M, 300, 300, 1, 0, 0);
    gemm(st, cx, T2, 300, 0, g.wh1, 300, 0, T4, 300, 0, g.bh1, M, 300, 300, 1, 1, 0);
    // out = tanh(h)*z + x*(1-z)
    k_ggnn_out<<<ewg(n), 256, 0, st>>>(T4, T1, X, out, n);
}

// ---------------------------------------------------------------------------
// kernel_launch
// ---------------------------------------------------------------------------
extern "C" void kernel_launch(void* const* d_in, const int* in_sizes, int n_in,
                              void* d_out, int out_size, void* d_ws, size_t ws_size,
                              hipStream_t stream)
{
    (void)in_sizes; (void)n_in; (void)out_size; (void)ws_size;
    const int B = 32, Lq = 30, E = 5, Ld = 100, Bd = 160, D = 300, IN = 300;
    const int WH = 5, EH = 5;

    const int*   query      = (const int*)  d_in[0];
    const float* query_adj  = (const float*)d_in[1];
    const float* query_mask = (const float*)d_in[2];
    const int*   doc        = (const int*)  d_in[3];
    const float* doc_adj    = (const float*)d_in[4];
    const float* doc_mask   = (const float*)d_in[5];
    const float* emb        = (const float*)d_in[6];

    auto ggp = [&](int base) {
        GG g;
        g.proj = (const float*)d_in[base + 0];
        g.wz0 = (const float*)d_in[base + 1];  g.bz0 = (const float*)d_in[base + 2];
        g.wz1 = (const float*)d_in[base + 3];  g.bz1 = (const float*)d_in[base + 4];
        g.wr0 = (const float*)d_in[base + 5];  g.br0 = (const float*)d_in[base + 6];
        g.wr1 = (const float*)d_in[base + 7];  g.br1 = (const float*)d_in[base + 8];
        g.wh0 = (const float*)d_in[base + 9];  g.bh0 = (const float*)d_in[base + 10];
        g.wh1 = (const float*)d_in[base + 11]; g.bh1 = (const float*)d_in[base + 12];
        return g;
    };
    const GG claim  = ggp(7);
    const GG fp1    = ggp(20);
    const GG scorer = ggp(33);
    const GG fp2    = ggp(46);
    const float* att_word_w1 = (const float*)d_in[59];  // (600, 300)
    const float* att_word_w2 = (const float*)d_in[60];  // (300, 5)
    const float* att_evd_w1  = (const float*)d_in[61];  // (1800, 300)
    const float* att_evd_w2  = (const float*)d_in[62];  // (300, 5)
    const float* mlp_w1      = (const float*)d_in[63];  // (7800, 300)
    const float* mlp_b1      = (const float*)d_in[64];
    const float* mlp_w2      = (const float*)d_in[65];  // (300, 2)
    const float* mlp_b2      = (const float*)d_in[66];

    // ---- workspace bump allocator (floats) ----
    float* ws = (float*)d_ws;
    size_t off = 0;
    auto alloc = [&](size_t nf) { float* p = ws + off; off += (nf + 63) & ~(size_t)63; return p; };

    const long long MB = (long long)Bd * Ld;          // 16000 doc rows
    float* eq   = alloc((size_t)B * Lq * IN);
    float* qh   = alloc((size_t)B * Lq * D);
    float* ed   = alloc((size_t)MB * IN);             // reused as docf later
    float* f1   = alloc((size_t)MB * D);
    float* X    = alloc((size_t)MB * D);
    float* Ag   = alloc((size_t)MB * D);
    float* T1   = alloc((size_t)MB * D);
    float* T2   = alloc((size_t)MB * D);
    float* T4   = alloc((size_t)MB * D);
    float* adjr = alloc((size_t)Bd * Ld * Ld);
    float* s_x  = alloc((size_t)MB);
    float* s_a  = alloc((size_t)MB);
    float* scor = alloc((size_t)MB);
    float* qr     = alloc((size_t)B * D);
    float* qpart  = alloc((size_t)B * D);
    float* lg1    = alloc((size_t)MB * WH);
    float* wsm1   = alloc((size_t)MB * WH);
    float* avg    = alloc((size_t)Bd * D * WH);       // (160, 1500)
    float* qpart2 = alloc((size_t)B * D);
    float* epart  = alloc((size_t)Bd * D);
    float* lg2    = alloc((size_t)Bd * EH);
    float* wsm2   = alloc((size_t)Bd * EH);
    float* att2   = alloc((size_t)B * D * WH * EH);   // (32, 7500)
    float* feat   = alloc((size_t)B * (D * WH * EH + D)); // (32, 7800)
    float* hid    = alloc((size_t)B * D);
    // packed bf16 operand scratch (max: A 16000x320, B 160 x 300 x 128)
    GemmCtx cx;
    cx.Ap = (unsigned short*)alloc((size_t)2700000);  // 5.4M ushorts
    cx.Bp = (unsigned short*)alloc((size_t)3200000);  // 6.4M ushorts

    // ---- claim branch ----
    k_gather<<<ewg((long long)B * Lq * IN), 256, 0, stream>>>(query, emb, eq, B * Lq, IN);
    ggnn_big(stream, cx, claim, query_adj, B, Lq, eq, IN, X, Ag, T1, T2, T4, qh);
    k_masked_mean<<<ewg((long long)B * D), 256, 0, stream>>>(qh, query_mask, qr, B, Lq, D);

    // ---- doc branch: feat_prop1 ----
    k_gather<<<ewg(MB * IN), 256, 0, stream>>>(doc, emb, ed, MB, IN);
    ggnn_big(stream, cx, fp1, doc_adj, Bd, Ld, ed, IN, X, Ag, T1, T2, T4, f1);

    // ---- word scorer (dout = 1) + GSL ----
    gemm(stream, cx, f1, D, 0, scorer.proj, 1, 0, s_x, 1, 0, nullptr, (int)MB, 1, D, 1, 0, 0);
    k_adj_matvec<<<ewg(MB), 256, 0, stream>>>(doc_adj, s_x, s_a, Bd, Ld);
    k_scorer_gates<<<ewg(MB), 256, 0, stream>>>(s_a, s_x,
        scorer.wz0, scorer.bz0, scorer.wz1, scorer.bz1,
        scorer.wr0, scorer.br0, scorer.wr1, scorer.br1,
        scorer.wh0, scorer.bh0, scorer.wh1, scorer.bh1, scor, (int)MB);
    k_gsl<<<Bd, 128, 0, stream>>>(doc_adj, scor, adjr, Ld, (int)(0.6f * Ld));

    // ---- feat_prop2 on refined graph (docf overwrites ed, which is dead) ----
    float* docf = ed;
    ggnn_big(stream, cx, fp2, adjr, Bd, Ld, f1, D, X, Ag, T1, T2, T4, docf);

    // ---- word attention ----
    gemm(stream, cx, qr, D, 0, att_word_w1, D, 0, qpart, D, 0, nullptr, B, D, D, 1, 0, 0);
    gemm(stream, cx, docf, D, 0, att_word_w1 + (size_t)D * D, D, 0, T1, D, 0,
         nullptr, (int)MB, D, D, 1, 0, 0);
    k_add_tanh_bcast<<<ewg(MB * D), 256, 0, stream>>>(T1, qpart, D, E * Ld, MB * D);
    gemm(stream, cx, T1, D, 0, att_word_w2, WH, 0, lg1, WH, 0, nullptr, (int)MB, WH, D, 1, 0, 0);
    k_softmax_axis<<<ewg((long long)Bd * WH), 256, 0, stream>>>(lg1, doc_mask, wsm1, Bd, Ld, WH);
    k_attend<<<ewg((long long)Bd * D * WH), 256, 0, stream>>>(docf, wsm1, avg, Bd, Ld, D, WH);

    // ---- evidence attention ----
    const int DW = D * WH;                  // 1500
    gemm(stream, cx, qr, D, 0, att_evd_w1, D, 0, qpart2, D, 0, nullptr, B, D, D, 1, 0, 0);
    gemm(stream, cx, avg, DW, 0, att_evd_w1 + (size_t)D * D, D, 0, epart, D, 0,
         nullptr, Bd, D, DW, 1, 0, 0);
    k_add_tanh_bcast<<<ewg((long long)Bd * D), 256, 0, stream>>>(epart, qpart2, D, E, (long long)Bd * D);
    gemm(stream, cx, epart, D, 0, att_evd_w2, EH, 0, lg2, EH, 0, nullptr, Bd, EH, D, 1, 0, 0);
    k_softmax_axis<<<ewg((long long)B * EH), 256, 0, stream>>>(lg2, nullptr, wsm2, B, E, EH);
    k_attend<<<ewg((long long)B * DW * EH), 256, 0, stream>>>(avg, wsm2, att2, B, E, DW, EH);

    // ---- MLP head ----
    const int DF = DW * EH;                 // 7500
    k_concat<<<ewg((long long)B * (DF + D)), 256, 0, stream>>>(att2, DF, qr, D, feat, B);
    gemm(stream, cx, feat, DF + D, 0, mlp_w1, D, 0, hid, D, 0, mlp_b1,
         B, D, DF + D, 1, 0, /*relu*/1);
    k_matmul_small<<<1, 64, 0, stream>>>(hid, mlp_w2, mlp_b2, (float*)d_out, B, D, 2);
}
